// Attention_1812476199118
// MI455X (gfx1250) — matrix operand order repaired and checked by reading, not compiled, not run on hardware
//
#include <hip/hip_runtime.h>

#define DIMC   384
#define KEYD   32
#define HEADS  8
#define NH_KD  256
#define DD     128
#define DHC    1024
#define H_QKV  1536
#define NPIX   196
#define NP     208     // N padded to multiple of 16
#define MP     224     // M (softmax axis) padded to multiple of 32 (K of A*V GEMM)
#define BATCH  256
#define RES    14
#define NGLOB  (BATCH * NP)   // flattened (batch, pixel) N dimension = 53248
#define EPSX   1e-5f
#define QSCALE 0.17677669529663689f  // 32^-0.5

typedef __bf16 bf16;
typedef __attribute__((ext_vector_type(16))) __bf16 v16bf;
typedef __attribute__((ext_vector_type(8)))  __bf16 v8bf;
typedef __attribute__((ext_vector_type(8)))  float  v8f;
typedef __attribute__((ext_vector_type(4))) unsigned int u32x4;
typedef __attribute__((ext_vector_type(8))) int          i32x8;
typedef __attribute__((ext_vector_type(4))) int          i32x4;

// ---- WMMA fragment loaders (CDNA5 16x16x32 bf16 layouts, wave32) ----
// A (16x32, row-major, lda in elems): lane L<16 -> row m0+L holds K {0..7,16..23};
// lanes 16..31 hold K {8..15,24..31} of row m0+(L&15).
__device__ __forceinline__ v16bf load_a_frag(const bf16* A, int lda,
                                             int m0, int k0, int lane) {
  int row  = m0 + (lane & 15);
  int koff = (lane >> 4) * 8;
  const bf16* p = A + (size_t)row * lda + k0 + koff;
  v8bf lo = *(const v8bf*)p;          // K = koff .. koff+7
  v8bf hi = *(const v8bf*)(p + 16);   // K = 16+koff .. 16+koff+7
  return __builtin_shufflevector(lo, hi, 0,1,2,3,4,5,6,7,8,9,10,11,12,13,14,15);
}
// B (32x16) supplied as BT[n][k] (K contiguous, ldb in elems): lane L holds
// column n0+(L&15); lanes 0-15 cover K {0..7,16..23}, lanes 16-31 cover K {8..15,24..31}.
__device__ __forceinline__ v16bf load_b_frag(const bf16* __restrict__ BT, int ldb,
                                             int n0, int k0, int lane) {
  int col  = n0 + (lane & 15);
  int koff = (lane >> 4) * 8;
  const bf16* p = BT + (size_t)col * ldb + k0 + koff;
  v8bf lo = *(const v8bf*)p;
  v8bf hi = *(const v8bf*)(p + 16);
  return __builtin_shufflevector(lo, hi, 0,1,2,3,4,5,6,7,8,9,10,11,12,13,14,15);
}
__device__ __forceinline__ v8f wmma_bf16(v16bf a, v16bf b, v8f c) {
  return __builtin_amdgcn_wmma_f32_16x16x32_bf16(false, a, false, b, (short)0, c,
                                                 false, false);
}

// ---- pack kernels ----
__global__ void pack_x_kernel(const float* __restrict__ x, bf16* __restrict__ xT) {
  int idx = blockIdx.x * 256 + threadIdx.x;
  if (idx >= BATCH * NP * DIMC) return;
  int c = idx % DIMC;
  int n = (idx / DIMC) % NP;
  int b = idx / (DIMC * NP);
  float v = (n < NPIX) ? x[((size_t)b * DIMC + c) * NPIX + n] : 0.f;
  xT[idx] = (bf16)v;
}

__global__ void pack_w_kernel(const float* __restrict__ w, bf16* __restrict__ wb, int count) {
  int idx = blockIdx.x * 256 + threadIdx.x;
  if (idx < count) wb[idx] = (bf16)w[idx];
}

// ---- stage 1: QKV GEMM + BN; weight tile staged to LDS by the Tensor Data
// Mover (TENSOR_LOAD_TO_LDS, TENSORcnt); A-frags then come from LDS, B streams
// from global. N flattened over (b, n); 2x4 tiles per wave. ----
__global__ void qkv_gemm_kernel(const bf16* __restrict__ wq, const bf16* __restrict__ xT,
                                const float* __restrict__ g, const float* __restrict__ bb,
                                const float* __restrict__ mmn, const float* __restrict__ vvr,
                                float* __restrict__ q_pre, bf16* __restrict__ k_t,
                                bf16* __restrict__ v_t) {
  extern __shared__ __align__(16) unsigned char smem_raw[];  // 128 x 384 bf16 = 96 KB
  bf16* Wlds = (bf16*)smem_raw;

  int lane = threadIdx.x & 31;
  int wave = threadIdx.x >> 5;
  int mw = wave & 3, nw = wave >> 2;
  int mbase = blockIdx.y * 128;
  int m0l   = mw * 32;                     // local LDS row of this wave's 2 M tiles
  int col0  = blockIdx.x * 128 + nw * 64;  // 4 N tiles (global (b,n) rows)

  if (wave == 0) {
    // TDM descriptor: 2D tensor, element 2B, tile 384(K) x 128(rows), row
    // stride 384 elems; destination = LDS offset 0 of this workgroup.
    unsigned long long gaddr =
        (unsigned long long)(uintptr_t)wq + (unsigned long long)mbase * DIMC * 2ull;
    unsigned lds_off = (unsigned)__builtin_amdgcn_groupstaticsize();
    u32x4 g0 = { 1u,                                   // count=1, is_restore=0
                 lds_off,                              // lds_addr
                 (unsigned)(gaddr & 0xFFFFFFFFull),    // global_addr[31:0]
                 ((unsigned)((gaddr >> 32) & 0x1FFFFFFull)) | 0x80000000u }; // addr hi | type=2
    i32x8 g1 = { (int)(1u << 16),                      // workgroup_mask=0, data_size=2B
                 (int)((DIMC & 0xFFFFu) << 16),        // tensor_dim0 = 384 (lo 16)
                 (int)((128u) << 16),                  // dim0 hi=0 | tensor_dim1 = 128 (lo 16)
                 (int)((DIMC & 0xFFFFu) << 16),        // dim1 hi=0 | tile_dim0 = 384
                 (int)128,                             // tile_dim1 = 128, tile_dim2 = 0
                 (int)DIMC,                            // tensor_dim0_stride = 384 (lo 32)
                 0, 0 };                               // stride0 hi, stride1 = 0
    i32x4 gz = { 0, 0, 0, 0 };
#if defined(__clang_major__) && __clang_major__ >= 23
    i32x8 gz8 = { 0, 0, 0, 0, 0, 0, 0, 0 };
    __builtin_amdgcn_tensor_load_to_lds(g0, g1, gz, gz, gz8, 0);
#else
    __builtin_amdgcn_tensor_load_to_lds(g0, g1, gz, gz, 0);
#endif
    __builtin_amdgcn_s_wait_tensorcnt(0);
  }
  __syncthreads();

  v8f acc[2][4] = {};
  for (int k0 = 0; k0 < DIMC; k0 += 32) {
    v16bf a0 = load_a_frag(Wlds, DIMC, m0l,      k0, lane);   // ds_load_b128
    v16bf a1 = load_a_frag(Wlds, DIMC, m0l + 16, k0, lane);
    __builtin_prefetch(xT + (size_t)(col0 + (lane & 15)) * DIMC + k0 + 64, 0, 1);
#pragma unroll
    for (int j = 0; j < 4; ++j) {
      v16bf bf = load_b_frag(xT, DIMC, col0 + 16 * j, k0, lane);
      acc[0][j] = wmma_bf16(a0, bf, acc[0][j]);
      acc[1][j] = wmma_bf16(a1, bf, acc[1][j]);
    }
  }
  int m0 = mbase + m0l;
  int rb = (lane >> 4) * 8;
  float sc[2][8], sh[2][8];
#pragma unroll
  for (int i = 0; i < 2; ++i)
#pragma unroll
    for (int r = 0; r < 8; ++r) {
      int o = m0 + 16 * i + rb + r;
      float s = g[o] * rsqrtf(vvr[o] + EPSX);
      sc[i][r] = s;
      sh[i][r] = bb[o] - mmn[o] * s;
    }
#pragma unroll
  for (int j = 0; j < 4; ++j) {
    int gi = col0 + 16 * j + (lane & 15);
    int b = gi / NP, n = gi - b * NP;
#pragma unroll
    for (int i = 0; i < 2; ++i) {
#pragma unroll
      for (int r = 0; r < 8; ++r) {
        int o = m0 + 16 * i + rb + r;
        float val = acc[i][j][r] * sc[i][r] + sh[i][r];
        if (o < NH_KD) {
          if (n < NPIX) q_pre[((size_t)b * NH_KD + o) * NPIX + n] = val;
        } else if (o < 2 * NH_KD) {
          int c = o - NH_KD; int h = c >> 5; int d = c & 31;
          k_t[(((size_t)b * HEADS + h) * NP + n) * KEYD + d] = (bf16)val;
        } else {
          int c = o - 2 * NH_KD; int h = c >> 7; int d = c & 127;
          v_t[(((size_t)b * HEADS + h) * DD + d) * MP + n] = (bf16)val;
        }
      }
    }
  }
}

// ---- stage 2: depthwise 3x3 + BN -> q_t (A-side layout for attention) ----
__global__ void dwconv_kernel(const float* __restrict__ q_pre, const float* __restrict__ w,
                              const float* __restrict__ g, const float* __restrict__ bb,
                              const float* __restrict__ mmn, const float* __restrict__ vvr,
                              bf16* __restrict__ q_t) {
  int idx = blockIdx.x * 256 + threadIdx.x;
  if (idx >= BATCH * NH_KD * NPIX) return;
  int n = idx % NPIX;
  int c = (idx / NPIX) % NH_KD;
  int b = idx / (NPIX * NH_KD);
  int y = n / RES, x = n % RES;
  const float* src = q_pre + ((size_t)b * NH_KD + c) * NPIX;
  float sum = 0.f;
  for (int ky = 0; ky < 3; ++ky)
    for (int kx = 0; kx < 3; ++kx) {
      int yy = y + ky - 1, xx = x + kx - 1;
      if (yy >= 0 && yy < RES && xx >= 0 && xx < RES)
        sum += src[yy * RES + xx] * w[c * 9 + ky * 3 + kx];
    }
  float s = g[c] * rsqrtf(vvr[c] + EPSX);
  float val = sum * s + (bb[c] - mmn[c] * s);
  int h = c >> 5, d = c & 31;
  q_t[(((size_t)b * HEADS + h) * NP + n) * KEYD + d] = (bf16)val;
}

// ---- stage 3: attn = Q^T K * scale + gathered bias; 1 A-frag x 4 B tiles ----
__global__ void qk_gemm_kernel(const bf16* __restrict__ q_t, const bf16* __restrict__ k_t,
                               const float* __restrict__ biases, const int* __restrict__ bidx,
                               float* __restrict__ attn_raw) {
  int lane = threadIdx.x & 31;
  int wave = threadIdx.x >> 5;
  int bh = blockIdx.z;
  int h = bh & 7;
  int nb = blockIdx.y * 8; if (nb > 13 - 8) nb = 13 - 8;  // clamped (overlap dup, deterministic)
  int mb = blockIdx.x * 4; if (mb > 13 - 4) mb = 13 - 4;
  int ntile = nb + wave;
  const bf16* A  = q_t + (size_t)bh * NP * KEYD;
  const bf16* Bt = k_t + (size_t)bh * NP * KEYD;
  v16bf a = load_a_frag(A, KEYD, ntile * 16, 0, lane);
  v8f acc[4] = {};
#pragma unroll
  for (int j = 0; j < 4; ++j) {
    v16bf bf = load_b_frag(Bt, KEYD, (mb + j) * 16, 0, lane);
    acc[j] = wmma_bf16(a, bf, acc[j]);
  }
  int rb = (lane >> 4) * 8;
#pragma unroll
  for (int j = 0; j < 4; ++j) {
    int m = (mb + j) * 16 + (lane & 15);
#pragma unroll
    for (int r = 0; r < 8; ++r) {
      int n = ntile * 16 + rb + r;
      float val = acc[j][r] * QSCALE;
      if (n < NPIX && m < NPIX) val += biases[h * NPIX + bidx[n * NPIX + m]];
      attn_raw[((size_t)bh * NP + n) * NP + m] = val;
    }
  }
}

// ---- stage 4: softmax, one wave32 per row; zero-fills padding m=196..223 ----
__global__ void softmax_kernel(const float* __restrict__ raw, bf16* __restrict__ attn_bf) {
  int lane = threadIdx.x;
  int row = blockIdx.x * blockDim.y + threadIdx.y;
  if (row >= BATCH * HEADS * NPIX) return;
  int bh = row / NPIX, n = row % NPIX;
  const float* rp = raw + ((size_t)bh * NP + n) * NP;
  float vals[7];
  float mx = -1e30f;
  for (int i = 0; i < 7; ++i) {
    int m = lane + i * 32;
    float v = (m < NPIX) ? rp[m] : -1e30f;
    vals[i] = v;
    mx = fmaxf(mx, v);
  }
  for (int off = 16; off >= 1; off >>= 1) mx = fmaxf(mx, __shfl_xor(mx, off, 32));
  float sum = 0.f;
  for (int i = 0; i < 7; ++i) { vals[i] = expf(vals[i] - mx); sum += vals[i]; }
  for (int off = 16; off >= 1; off >>= 1) sum += __shfl_xor(sum, off, 32);
  float inv = 1.f / sum;
  bf16* op = attn_bf + ((size_t)bh * NP + n) * MP;
  for (int i = 0; i < 7; ++i) {
    int m = lane + i * 32;                  // covers exactly MP = 224
    op[m] = (bf16)((m < NPIX) ? vals[i] * inv : 0.f);
  }
}

// ---- stage 5: out = V * attn^T, fused ReLU; 2x2 tiles per wave ----
__global__ void av_gemm_kernel(const bf16* __restrict__ v_t, const bf16* __restrict__ attn_bf,
                               bf16* __restrict__ avT) {
  int lane = threadIdx.x & 31;
  int wave = threadIdx.x >> 5;
  int mw = wave & 3, nw = wave >> 2;
  int bh = blockIdx.y;
  int b = bh >> 3, h = bh & 7;
  int base = blockIdx.x * 64; if (base > NP - 64) base = NP - 64;  // clamped overlap
  int m0   = mw * 32;            // 2 M tiles cover d = 0..127 across 4 waves
  int col0 = base + nw * 32;     // 2 N tiles
  const bf16* A  = v_t + (size_t)bh * DD * MP;
  const bf16* Bt = attn_bf + (size_t)bh * NP * MP;
  v8f acc[2][2] = {};
  for (int k0 = 0; k0 < MP; k0 += 32) {
    v16bf a0 = load_a_frag(A, MP, m0,      k0, lane);
    v16bf a1 = load_a_frag(A, MP, m0 + 16, k0, lane);
#pragma unroll
    for (int j = 0; j < 2; ++j) {
      v16bf bf = load_b_frag(Bt, MP, col0 + 16 * j, k0, lane);
      acc[0][j] = wmma_bf16(a0, bf, acc[0][j]);
      acc[1][j] = wmma_bf16(a1, bf, acc[1][j]);
    }
  }
  int rb = (lane >> 4) * 8;
#pragma unroll
  for (int j = 0; j < 2; ++j) {
    int n = col0 + 16 * j + (lane & 15);
#pragma unroll
    for (int i = 0; i < 2; ++i) {
      v8bf tmp;
#pragma unroll
      for (int r = 0; r < 8; ++r) tmp[r] = (bf16)fmaxf(acc[i][j][r], 0.f);
      // 8 accumulator rows per lane are consecutive channels -> one 16B store
      *(v8bf*)(avT + (((size_t)b * NP + n) * DHC + h * DD + m0 + 16 * i + rb)) = tmp;
    }
  }
}

// ---- stage 6: proj GEMM + BN -> f32 output; N flattened over (b, n) ----
__global__ void proj_gemm_kernel(const bf16* __restrict__ wp, const bf16* __restrict__ avT,
                                 const float* __restrict__ g, const float* __restrict__ bb,
                                 const float* __restrict__ mmn, const float* __restrict__ vvr,
                                 float* __restrict__ out) {
  int lane = threadIdx.x & 31;
  int wave = threadIdx.x >> 5;
  int mw = wave & 3, nw = wave >> 2;
  int m0   = blockIdx.y * 128 + mw * 32;
  int col0 = blockIdx.x * 128 + nw * 64;
  v8f acc[2][4] = {};
  for (int k0 = 0; k0 < DHC; k0 += 32) {
    v16bf a0 = load_a_frag(wp, DHC, m0,      k0, lane);
    v16bf a1 = load_a_frag(wp, DHC, m0 + 16, k0, lane);
    __builtin_prefetch(wp + (size_t)(m0 + (lane & 15)) * DHC + k0 + 64, 0, 1);
#pragma unroll
    for (int j = 0; j < 4; ++j) {
      v16bf bf = load_b_frag(avT, DHC, col0 + 16 * j, k0, lane);
      acc[0][j] = wmma_bf16(a0, bf, acc[0][j]);
      acc[1][j] = wmma_bf16(a1, bf, acc[1][j]);
    }
  }
  int rb = (lane >> 4) * 8;
  float sc[2][8], sh[2][8];
#pragma unroll
  for (int i = 0; i < 2; ++i)
#pragma unroll
    for (int r = 0; r < 8; ++r) {
      int o = m0 + 16 * i + rb + r;
      float s = g[o] * rsqrtf(vvr[o] + EPSX);
      sc[i][r] = s;
      sh[i][r] = bb[o] - mmn[o] * s;
    }
#pragma unroll
  for (int j = 0; j < 4; ++j) {
    int gi = col0 + 16 * j + (lane & 15);
    int b = gi / NP, n = gi - b * NP;
    if (n >= NPIX) continue;
#pragma unroll
    for (int i = 0; i < 2; ++i)
#pragma unroll
      for (int r = 0; r < 8; ++r) {
        int o = m0 + 16 * i + rb + r;
        out[((size_t)b * DIMC + o) * NPIX + n] = acc[i][j][r] * sc[i][r] + sh[i][r];
      }
  }
}

extern "C" void kernel_launch(void* const* d_in, const int* in_sizes, int n_in,
                              void* d_out, int out_size, void* d_ws, size_t ws_size,
                              hipStream_t stream) {
  const float* x      = (const float*)d_in[0];
  const float* qkv_w  = (const float*)d_in[1];
  const float* qkv_g  = (const float*)d_in[2];
  const float* qkv_b  = (const float*)d_in[3];
  const float* qkv_m  = (const float*)d_in[4];
  const float* qkv_v  = (const float*)d_in[5];
  const float* dw_w   = (const float*)d_in[6];
  const float* dw_g   = (const float*)d_in[7];
  const float* dw_b   = (const float*)d_in[8];
  const float* dw_m   = (const float*)d_in[9];
  const float* dw_v   = (const float*)d_in[10];
  const float* proj_w = (const float*)d_in[11];
  const float* proj_g = (const float*)d_in[12];
  const float* proj_b = (const float*)d_in[13];
  const float* proj_m = (const float*)d_in[14];
  const float* proj_v = (const float*)d_in[15];
  const float* ab     = (const float*)d_in[16];
  const int*   bidx   = (const int*)d_in[17];
  float* out = (float*)d_out;

  char* ws = (char*)d_ws;
  size_t off = 0;
  auto alloc = [&](size_t bytes) -> char* {
    char* p = ws + off;
    off += (bytes + 255) & ~(size_t)255;
    return p;
  };
  bf16*  xT       = (bf16*) alloc((size_t)BATCH * NP * DIMC * 2);
  bf16*  wqkv     = (bf16*) alloc((size_t)H_QKV * DIMC * 2);
  bf16*  wproj    = (bf16*) alloc((size_t)DIMC * DHC * 2);
  float* q_pre    = (float*)alloc((size_t)BATCH * NH_KD * NPIX * 4);
  bf16*  q_t      = (bf16*) alloc((size_t)BATCH * HEADS * NP * KEYD * 2);
  bf16*  k_t      = (bf16*) alloc((size_t)BATCH * HEADS * NP * KEYD * 2);
  bf16*  v_t      = (bf16*) alloc((size_t)BATCH * HEADS * DD * MP * 2);
  float* attn_raw = (float*)alloc((size_t)BATCH * HEADS * NP * NP * 4);
  bf16*  attn_bf  = (bf16*) alloc((size_t)BATCH * HEADS * NP * MP * 2);
  bf16*  avT      = (bf16*) alloc((size_t)BATCH * NP * DHC * 2);

  pack_x_kernel<<<(BATCH * NP * DIMC + 255) / 256, 256, 0, stream>>>(x, xT);
  pack_w_kernel<<<(H_QKV * DIMC + 255) / 256, 256, 0, stream>>>(qkv_w, wqkv, H_QKV * DIMC);
  pack_w_kernel<<<(DIMC * DHC + 255) / 256, 256, 0, stream>>>(proj_w, wproj, DIMC * DHC);

  // N flattened over (b, n): NGLOB/128 = 416 blocks; M: 1536/128 = 12.
  // 96 KB dynamic LDS for the TDM-staged weight tile (320 KB/WGP available).
  qkv_gemm_kernel<<<dim3(NGLOB / 128, H_QKV / 128), 256, 128 * DIMC * 2, stream>>>(
      wqkv, xT, qkv_g, qkv_b, qkv_m, qkv_v, q_pre, k_t, v_t);

  dwconv_kernel<<<(BATCH * NH_KD * NPIX + 255) / 256, 256, 0, stream>>>(
      q_pre, dw_w, dw_g, dw_b, dw_m, dw_v, q_t);

  qk_gemm_kernel<<<dim3(4, 2, BATCH * HEADS), 256, 0, stream>>>(
      q_t, k_t, ab, bidx, attn_raw);

  softmax_kernel<<<dim3((BATCH * HEADS * NPIX + 7) / 8), dim3(32, 8), 0, stream>>>(
      attn_raw, attn_bf);

  av_gemm_kernel<<<dim3(4, BATCH * HEADS), 256, 0, stream>>>(v_t, attn_bf, avT);

  proj_gemm_kernel<<<dim3(NGLOB / 128, DIMC / 128), 256, 0, stream>>>(
      wproj, avT, proj_g, proj_b, proj_m, proj_v, out);
}
